// DiffCSPNet_45973329936680
// MI455X (gfx1250) — matrix operand order repaired and checked by reading, not compiled, not run on hardware
//
#include <hip/hip_runtime.h>
#include <math.h>

typedef __attribute__((ext_vector_type(8)))  __bf16 v8bf;
typedef __attribute__((ext_vector_type(16))) __bf16 v16bf;
typedef __attribute__((ext_vector_type(8)))  float  v8f;

__device__ __forceinline__ __bf16 f2bf(float f) {
    unsigned u = __builtin_bit_cast(unsigned, f);
    unsigned r = (u + 0x7FFFu + ((u >> 16) & 1u)) >> 16;
    unsigned short h = (unsigned short)r;
    return __builtin_bit_cast(__bf16, h);
}
__device__ __forceinline__ float silu_f(float v) { return v / (1.0f + __expf(-v)); }

#define WMMA_BF16(a, b, c) \
    __builtin_amdgcn_wmma_f32_16x16x32_bf16(false, (a), false, (b), (short)0, (c), false, false)

// ---------------------------------------------------------------------------
// LayerNorm: one block (256 threads) per node row, H == 256. Writes bf16 h.
// ---------------------------------------------------------------------------
__global__ void ln_kernel(const float* __restrict__ x,
                          const float* __restrict__ gamma,
                          const float* __restrict__ beta,
                          __bf16* __restrict__ hbf, int N) {
    int row = blockIdx.x;
    int t = threadIdx.x;
    float v = x[(size_t)row * 256 + t];
    float s = v, s2 = v * v;
    for (int o = 16; o > 0; o >>= 1) {
        s  += __shfl_xor(s,  o, 32);
        s2 += __shfl_xor(s2, o, 32);
    }
    __shared__ float red0[8], red1[8];
    int lane = t & 31, wave = t >> 5;
    if (lane == 0) { red0[wave] = s; red1[wave] = s2; }
    __syncthreads();
    float ts = 0.f, ts2 = 0.f;
    for (int i = 0; i < 8; ++i) { ts += red0[i]; ts2 += red1[i]; }
    float mean = ts * (1.0f / 256.0f);
    float var  = ts2 * (1.0f / 256.0f) - mean * mean;
    float rstd = rsqrtf(var + 1e-5f);
    float hv = (v - mean) * rstd * gamma[t] + beta[t];
    hbf[(size_t)row * 256 + t] = f2bf(hv);
}

// ---------------------------------------------------------------------------
// Lattice products: ltl[g, i*3+k] = sum_j L[g,i,j] * L[g,k,j]  (bf16 out)
// ---------------------------------------------------------------------------
__global__ void ltl_kernel(const float* __restrict__ L, __bf16* __restrict__ ltl, int G) {
    int idx = blockIdx.x * 256 + threadIdx.x;
    if (idx >= G * 9) return;
    int g = idx / 9, ij = idx % 9;
    int i = ij / 3, k = ij % 3;
    const float* Lg = L + (size_t)g * 9;
    float s = Lg[i*3+0]*Lg[k*3+0] + Lg[i*3+1]*Lg[k*3+1] + Lg[i*3+2]*Lg[k*3+2];
    ltl[idx] = f2bf(s);
}

// ---------------------------------------------------------------------------
// Pack an f32 [Kreal, 256] weight into bf16 WMMA B-operand tiles:
// layout [kt][nt (16)][lane (32)][i (16)], k = kt*32 + (i&7) + ((lane&16)>>1) + ((i&8)<<1),
// n = nt*16 + (lane&15). Each lane's 16 bf16 are contiguous (32B load).
// ---------------------------------------------------------------------------
__global__ void pack_kernel(const float* __restrict__ W, __bf16* __restrict__ out,
                            int Kreal, int total) {
    int idx = blockIdx.x * 256 + threadIdx.x;
    if (idx >= total) return;
    int i    = idx & 15;
    int lane = (idx >> 4) & 31;
    int nt   = (idx >> 9) & 15;
    int kt   = idx >> 13;
    int k = kt * 32 + (i & 7) + ((lane & 16) >> 1) + ((i & 8) << 1);
    int n = nt * 16 + (lane & 15);
    float v = (k < Kreal) ? W[(size_t)k * 256 + n] : 0.0f;
    out[idx] = f2bf(v);
}

// ---------------------------------------------------------------------------
// A operand from LDS row-major row pointer:
// lane<16: k-chunks {0..7,16..23}; lane>=16: {8..15,24..31}.
// ---------------------------------------------------------------------------
__device__ __forceinline__ v16bf load_a(const __bf16* rowp, int abase) {
    v8bf lo = *(const v8bf*)(rowp + abase);
    v8bf hi = *(const v8bf*)(rowp + abase + 16);
    return __builtin_shufflevector(lo, hi, 0,1,2,3,4,5,6,7,8,9,10,11,12,13,14,15);
}

// ---------------------------------------------------------------------------
// Edge kernel: 64 edges per block, 256 threads (8 waves).
// Each wave owns a 32-column strip (2 N-tiles) and 4 M-sub-tiles (64 rows):
// B tiles are loaded once per K-step and reused by 4 WMMAs each.
// x = [hi(256) | hj(256) | ltl(9) | sin/cos(384) | pad->928] in LDS bf16.
// ---------------------------------------------------------------------------
__global__ void __launch_bounds__(256)
edge_kernel(const __bf16* __restrict__ hbf,
            const __bf16* __restrict__ ltl,
            const int*    __restrict__ eidx,   // [2, E]
            const int*    __restrict__ e2g,    // [E]
            const float*  __restrict__ fdp,    // [E, 3]
            const __bf16* __restrict__ W1p,    // [KT1][16][32][16]
            const float*  __restrict__ b1,
            const __bf16* __restrict__ W2p,    // [8][16][32][16]
            const float*  __restrict__ b2,
            float* __restrict__ sums,
            float* __restrict__ cnt,
            int E, int KT1) {
    __shared__ __attribute__((aligned(16))) __bf16 sx[64 * 928];   // 116 KB
    __shared__ __attribute__((aligned(16))) __bf16 se1[64 * 256];  // 32 KB
    __shared__ int ssrc[64], sdst[64];

    int t = threadIdx.x;
    int e0 = blockIdx.x * 64;

    if (t < 64) {
        int e = e0 + t; if (e >= E) e = E - 1;
        ssrc[t] = eidx[e];
    } else if (t < 128) {
        int r = t - 64;
        int e = e0 + r; if (e >= E) e = E - 1;
        sdst[r] = eidx[(size_t)E + e];
    }
    __syncthreads();

    const __bf16 zb = f2bf(0.0f);
    // hi / hj
    for (int idx = t; idx < 64 * 256; idx += 256) {
        int r = idx >> 8, c = idx & 255;
        bool ok = (e0 + r) < E;
        sx[r*928 + c]       = ok ? hbf[(size_t)ssrc[r]*256 + c] : zb;
        sx[r*928 + 256 + c] = ok ? hbf[(size_t)sdst[r]*256 + c] : zb;
    }
    // lattice (9) + tail pad (905..927)
    for (int idx = t; idx < 64 * 32; idx += 256) {
        int r = idx >> 5, c = idx & 31;
        int e = e0 + r;
        if (c < 9) {
            bool ok = e < E;
            int g = ok ? e2g[e] : 0;
            sx[r*928 + 512 + c] = ok ? ltl[(size_t)g*9 + c] : zb;
        } else {
            sx[r*928 + 905 + (c - 9)] = zb;  // c-9 in [0,23)
        }
    }
    // sinusoid embedding: cols 521..904
    for (int idx = t; idx < 64 * 384; idx += 256) {
        int r = idx / 384, j = idx - r * 384;
        int e = e0 + r;
        bool ok = e < E;
        int jj = (j < 192) ? j : (j - 192);
        int s  = jj >> 6, f = jj & 63;
        float xv = ok ? fdp[(size_t)e*3 + s] : 0.0f;
        float arg = xv * (6.2831853071795864f * (float)f);
        float v = (j < 192) ? __sinf(arg) : __cosf(arg);
        sx[r*928 + 521 + j] = ok ? f2bf(v) : zb;
    }
    __syncthreads();

    int lane = t & 31, wave = t >> 5;
    int abase = (lane & 16) ? 8 : 0;
    int arow  = lane & 15;

    // ---- GEMM1: [64 x 928] x [928 x 256] ----
    v8f acc[4][2] = {};
    for (int kt = 0; kt < KT1; ++kt) {
        const __bf16* bp = W1p + (((size_t)(kt * 16 + wave * 2) * 32 + lane) * 16);
        __builtin_prefetch(bp + 16 * 512, 0, 0);
        v16bf bv0 = *(const v16bf*)(bp);
        v16bf bv1 = *(const v16bf*)(bp + 512);
        #pragma unroll
        for (int m = 0; m < 4; ++m) {
            v16bf a = load_a(&sx[(m * 16 + arow) * 928 + kt * 32], abase);
            acc[m][0] = WMMA_BF16(a, bv0, acc[m][0]);
            acc[m][1] = WMMA_BF16(a, bv1, acc[m][1]);
        }
    }
    // bias + SiLU -> se1 (bf16)
    int crow = (lane & 16) ? 8 : 0;
    int ccol = wave * 32 + (lane & 15);
    float bb0 = b1[ccol], bb1 = b1[ccol + 16];
    #pragma unroll
    for (int m = 0; m < 4; ++m) {
        #pragma unroll
        for (int i = 0; i < 8; ++i) {
            int row = m * 16 + crow + i;
            se1[row*256 + ccol]      = f2bf(silu_f(acc[m][0][i] + bb0));
            se1[row*256 + ccol + 16] = f2bf(silu_f(acc[m][1][i] + bb1));
        }
    }
    __syncthreads();

    // ---- GEMM2: [64 x 256] x [256 x 256] ----
    #pragma unroll
    for (int m = 0; m < 4; ++m) { acc[m][0] = (v8f){}; acc[m][1] = (v8f){}; }
    for (int kt = 0; kt < 8; ++kt) {
        const __bf16* bp = W2p + (((size_t)(kt * 16 + wave * 2) * 32 + lane) * 16);
        v16bf bv0 = *(const v16bf*)(bp);
        v16bf bv1 = *(const v16bf*)(bp + 512);
        #pragma unroll
        for (int m = 0; m < 4; ++m) {
            v16bf a = load_a(&se1[(m * 16 + arow) * 256 + kt * 32], abase);
            acc[m][0] = WMMA_BF16(a, bv0, acc[m][0]);
            acc[m][1] = WMMA_BF16(a, bv1, acc[m][1]);
        }
    }
    float cb0 = b2[ccol], cb1 = b2[ccol + 16];
    #pragma unroll
    for (int m = 0; m < 4; ++m) {
        #pragma unroll
        for (int i = 0; i < 8; ++i) {
            int row = m * 16 + crow + i;
            if (e0 + row < E) {
                int sidx = ssrc[row];
                atomicAdd(&sums[(size_t)sidx * 256 + ccol],      silu_f(acc[m][0][i] + cb0));
                atomicAdd(&sums[(size_t)sidx * 256 + ccol + 16], silu_f(acc[m][1][i] + cb1));
            }
        }
    }
    if (t < 64 && (e0 + t) < E) atomicAdd(&cnt[ssrc[t]], 1.0f);
}

// ---------------------------------------------------------------------------
// Node kernel: 64 nodes per block. y = [h | agg] (64x512 bf16 in LDS),
// GEMM (K=512) -> SiLU -> GEMM (K=256) -> SiLU -> residual add.
// ---------------------------------------------------------------------------
__global__ void __launch_bounds__(256)
node_kernel(const float*  __restrict__ nf,
            const __bf16* __restrict__ hbf,
            const float*  __restrict__ sums,
            const float*  __restrict__ cnt,
            const __bf16* __restrict__ Wn1p,  // [16][16][32][16]
            const float*  __restrict__ bn1,
            const __bf16* __restrict__ Wn2p,  // [8][16][32][16]
            const float*  __restrict__ bn2,
            float* __restrict__ out,
            int N) {
    __shared__ __attribute__((aligned(16))) __bf16 sy[64 * 512];   // 64 KB
    __shared__ __attribute__((aligned(16))) __bf16 sz[64 * 256];   // 32 KB

    int t = threadIdx.x;
    int n0 = blockIdx.x * 64;

    const __bf16 zb = f2bf(0.0f);
    for (int idx = t; idx < 64 * 256; idx += 256) {
        int r = idx >> 8, c = idx & 255;
        int node = n0 + r;
        bool ok = node < N;
        int nc = ok ? node : 0;
        float ct = cnt[nc];
        float inv = 1.0f / fmaxf(ct, 1.0f);
        sy[r*512 + c]       = ok ? hbf[(size_t)nc*256 + c] : zb;
        sy[r*512 + 256 + c] = ok ? f2bf(sums[(size_t)nc*256 + c] * inv) : zb;
    }
    __syncthreads();

    int lane = t & 31, wave = t >> 5;
    int abase = (lane & 16) ? 8 : 0;
    int arow  = lane & 15;

    // ---- GEMM1: [64 x 512] x [512 x 256] ----
    v8f acc[4][2] = {};
    for (int kt = 0; kt < 16; ++kt) {
        const __bf16* bp = Wn1p + (((size_t)(kt * 16 + wave * 2) * 32 + lane) * 16);
        __builtin_prefetch(bp + 16 * 512, 0, 0);
        v16bf bv0 = *(const v16bf*)(bp);
        v16bf bv1 = *(const v16bf*)(bp + 512);
        #pragma unroll
        for (int m = 0; m < 4; ++m) {
            v16bf a = load_a(&sy[(m * 16 + arow) * 512 + kt * 32], abase);
            acc[m][0] = WMMA_BF16(a, bv0, acc[m][0]);
            acc[m][1] = WMMA_BF16(a, bv1, acc[m][1]);
        }
    }
    int crow = (lane & 16) ? 8 : 0;
    int ccol = wave * 32 + (lane & 15);
    float bb0 = bn1[ccol], bb1 = bn1[ccol + 16];
    #pragma unroll
    for (int m = 0; m < 4; ++m) {
        #pragma unroll
        for (int i = 0; i < 8; ++i) {
            int row = m * 16 + crow + i;
            sz[row*256 + ccol]      = f2bf(silu_f(acc[m][0][i] + bb0));
            sz[row*256 + ccol + 16] = f2bf(silu_f(acc[m][1][i] + bb1));
        }
    }
    __syncthreads();

    // ---- GEMM2: [64 x 256] x [256 x 256] ----
    #pragma unroll
    for (int m = 0; m < 4; ++m) { acc[m][0] = (v8f){}; acc[m][1] = (v8f){}; }
    for (int kt = 0; kt < 8; ++kt) {
        const __bf16* bp = Wn2p + (((size_t)(kt * 16 + wave * 2) * 32 + lane) * 16);
        v16bf bv0 = *(const v16bf*)(bp);
        v16bf bv1 = *(const v16bf*)(bp + 512);
        #pragma unroll
        for (int m = 0; m < 4; ++m) {
            v16bf a = load_a(&sz[(m * 16 + arow) * 256 + kt * 32], abase);
            acc[m][0] = WMMA_BF16(a, bv0, acc[m][0]);
            acc[m][1] = WMMA_BF16(a, bv1, acc[m][1]);
        }
    }
    float cb0 = bn2[ccol], cb1 = bn2[ccol + 16];
    #pragma unroll
    for (int m = 0; m < 4; ++m) {
        #pragma unroll
        for (int i = 0; i < 8; ++i) {
            int row = m * 16 + crow + i;
            int node = n0 + row;
            if (node < N) {
                size_t o0 = (size_t)node * 256 + ccol;
                out[o0]      = nf[o0]      + silu_f(acc[m][0][i] + cb0);
                out[o0 + 16] = nf[o0 + 16] + silu_f(acc[m][1][i] + cb1);
            }
        }
    }
}

// ---------------------------------------------------------------------------
extern "C" void kernel_launch(void* const* d_in, const int* in_sizes, int n_in,
                              void* d_out, int out_size, void* d_ws, size_t ws_size,
                              hipStream_t stream) {
    const float* nf      = (const float*)d_in[0];
    const float* latt    = (const float*)d_in[1];
    const int*   eidx    = (const int*)  d_in[2];
    const int*   e2g     = (const int*)  d_in[3];
    const float* fdp     = (const float*)d_in[4];
    const float* gamma   = (const float*)d_in[6];
    const float* beta    = (const float*)d_in[7];
    const float* We1     = (const float*)d_in[8];
    const float* be1     = (const float*)d_in[9];
    const float* We2     = (const float*)d_in[10];
    const float* be2     = (const float*)d_in[11];
    const float* Wn1     = (const float*)d_in[12];
    const float* bn1     = (const float*)d_in[13];
    const float* Wn2     = (const float*)d_in[14];
    const float* bn2     = (const float*)d_in[15];

    const int H   = in_sizes[6];            // 256
    const int N   = in_sizes[0] / H;        // 20000
    const int E   = in_sizes[3];            // 320000
    const int G   = in_sizes[5];            // 1000
    const int EIN = in_sizes[8] / H;        // 905
    const int KT1 = (EIN + 31) / 32;        // 29

    // workspace carve-out
    size_t off = 0;
    auto carve = [&](size_t bytes) { size_t o = off; off = (off + bytes + 255) & ~(size_t)255; return o; };
    char* ws = (char*)d_ws;
    __bf16* hbf   = (__bf16*)(ws + carve((size_t)N * 256 * 2));
    __bf16* ltl   = (__bf16*)(ws + carve((size_t)G * 9 * 2));
    float*  sums  = (float*) (ws + carve((size_t)N * 256 * 4));
    float*  cnt   = (float*) (ws + carve((size_t)N * 4));
    __bf16* W1p   = (__bf16*)(ws + carve((size_t)KT1 * 8192 * 2));
    __bf16* W2p   = (__bf16*)(ws + carve((size_t)8   * 8192 * 2));
    __bf16* Wn1p  = (__bf16*)(ws + carve((size_t)16  * 8192 * 2));
    __bf16* Wn2p  = (__bf16*)(ws + carve((size_t)8   * 8192 * 2));
    (void)ws_size; (void)n_in; (void)out_size;

    // preprocessing
    ln_kernel<<<N, 256, 0, stream>>>(nf, gamma, beta, hbf, N);
    ltl_kernel<<<(G * 9 + 255) / 256, 256, 0, stream>>>(latt, ltl, G);

    int tot1 = KT1 * 8192;
    pack_kernel<<<(tot1 + 255) / 256, 256, 0, stream>>>(We1, W1p, EIN, tot1);
    pack_kernel<<<(8 * 8192 + 255) / 256, 256, 0, stream>>>(We2, W2p, 256, 8 * 8192);
    pack_kernel<<<(16 * 8192 + 255) / 256, 256, 0, stream>>>(Wn1, Wn1p, 512, 16 * 8192);
    pack_kernel<<<(8 * 8192 + 255) / 256, 256, 0, stream>>>(Wn2, Wn2p, 256, 8 * 8192);

    hipMemsetAsync(sums, 0, (size_t)N * 256 * 4, stream);
    hipMemsetAsync(cnt, 0, (size_t)N * 4, stream);

    // edge MLP + scatter (64 edges / block)
    edge_kernel<<<(E + 63) / 64, 256, 0, stream>>>(hbf, ltl, eidx, e2g, fdp,
                                                   W1p, be1, W2p, be2,
                                                   sums, cnt, E, KT1);
    // node MLP + residual (64 nodes / block)
    node_kernel<<<(N + 63) / 64, 256, 0, stream>>>(nf, hbf, sums, cnt,
                                                   Wn1p, bn1, Wn2p, bn2,
                                                   (float*)d_out, N);
}
